// SMoEAdapter_84542136254977
// MI455X (gfx1250) — compile-verified, matrix operand order checked by reading, not run on hardware
//
#include <hip/hip_runtime.h>
#include <hip/hip_bf16.h>
#include <math.h>

// Problem sizes (fixed by the reference)
#define B_  16
#define S_  2048
#define D_  1024
#define H_  256
#define E_  128
#define BS_ (B_*S_)

typedef __attribute__((ext_vector_type(16))) __bf16 v16bf;
typedef __attribute__((ext_vector_type(8)))  float  v8f;

union FragBF {
    v16bf v;
    uint4 q[2];
    unsigned short u[16];
};

__device__ __forceinline__ unsigned short f32_to_bf16_rne(float f) {
    union { float f; unsigned u; } a; a.f = f;
    unsigned u = a.u;
    u += 0x7FFFu + ((u >> 16) & 1u);   // round-to-nearest-even
    return (unsigned short)(u >> 16);
}

// WMMA bf16 16x16x32 per-lane fragment (ISA 7.12.2, wave32):
// lane holds M(or N) = lane&15; lane-half kh=(lane>>4) selects K+8*kh.
// Half i -> K offset k(i) = ((i&8)<<1) + (i&7): two contiguous 8-half runs
// {0..7} and {16..23}.  With bf16 data pre-packed K-contiguously in memory,
// a fragment is exactly two 16-byte loads.
__device__ __forceinline__ v16bf load_frag_bf(const unsigned short* __restrict__ p) {
    FragBF f;
    f.q[0] = *(const uint4*)(p);        // K  0.. 7 (+8kh)
    f.q[1] = *(const uint4*)(p + 16);   // K 16..23 (+8kh)
    return f.v;
}

__device__ __forceinline__ v8f wmma_bf16(v16bf a, v16bf b, v8f c) {
    return __builtin_amdgcn_wmma_f32_16x16x32_bf16(
        false, a, false, b, (short)0, c, false, false);
}

// ================= conversion / transpose passes (bandwidth-bound) =================

// elementwise f32 -> bf16, 2 elems/thread (n must be even)
__global__ __launch_bounds__(256) void cvt_bf16_kernel(const float* __restrict__ src,
                                                       unsigned short* __restrict__ dst,
                                                       long n2) {
    long i = (long)blockIdx.x * 256 + threadIdx.x;
    if (i < n2) {
        float2 v = ((const float2*)src)[i];
        unsigned p = (unsigned)f32_to_bf16_rne(v.x) | ((unsigned)f32_to_bf16_rne(v.y) << 16);
        ((unsigned*)dst)[i] = p;
    }
}

// phi [H,E] f32 -> phiT [E,H] bf16
__global__ __launch_bounds__(256) void phiT_kernel(const float* __restrict__ phi,
                                                   unsigned short* __restrict__ phiT) {
    int idx = blockIdx.x * 256 + threadIdx.x;   // H_*E_ = 32768
    int h = idx >> 7;                           // /E_
    int e = idx & (E_ - 1);
    phiT[e * H_ + h] = f32_to_bf16_rne(phi[idx]);
}

// expert_w [E,H,H] f32 -> ewT [E,H,H] bf16 with last two dims swapped
__global__ __launch_bounds__(256) void ewT_kernel(const float* __restrict__ ew,
                                                  unsigned short* __restrict__ ewT) {
    long idx = (long)blockIdx.x * 256 + threadIdx.x;   // E_*H_*H_ = 8388608
    int f = (int)(idx & (H_ - 1));
    int h = (int)((idx >> 8) & (H_ - 1));
    int e = (int)(idx >> 16);
    ewT[((long)e * H_ + f) * H_ + h] = f32_to_bf16_rne(ew[idx]);
}

// ================= K1: xs = gelu(x @ fc1_w^T + fc1_b)  (WMMA, 2x4 tiles) =============
// writes xs_bf [BS,H] and xsT_bf [B,H,S]
__global__ __launch_bounds__(32) void fc1_gelu_kernel(const unsigned short* __restrict__ xb,
                                                      const unsigned short* __restrict__ w1b,
                                                      const float* __restrict__ b1,
                                                      unsigned short* __restrict__ xs_bf,
                                                      unsigned short* __restrict__ xsT_bf) {
    int ntile = blockIdx.x & 3;            // H/64 = 4
    int mtile = blockIdx.x >> 2;           // BS/32 = 1024
    int lane  = threadIdx.x;
    int l15 = lane & 15, kh8 = 8 * (lane >> 4);
    int m0 = mtile * 32, n0 = ntile * 64;
    v8f acc[2][4] = {};
    for (int k0 = 0; k0 < D_; k0 += 32) {
        v16bf a[2], b[4];
#pragma unroll
        for (int i = 0; i < 2; ++i)
            a[i] = load_frag_bf(xb + (long)(m0 + i * 16 + l15) * D_ + k0 + kh8);
#pragma unroll
        for (int j = 0; j < 4; ++j)
            b[j] = load_frag_bf(w1b + (long)(n0 + j * 16 + l15) * D_ + k0 + kh8);
#pragma unroll
        for (int i = 0; i < 2; ++i)
#pragma unroll
            for (int j = 0; j < 4; ++j)
                acc[i][j] = wmma_bf16(a[i], b[j], acc[i][j]);
    }
#pragma unroll
    for (int i = 0; i < 2; ++i)
#pragma unroll
        for (int j = 0; j < 4; ++j) {
            int n = n0 + j * 16 + l15;
            float bias = b1[n];
#pragma unroll
            for (int r = 0; r < 8; ++r) {
                int m = m0 + i * 16 + kh8 + r;
                float v = acc[i][j][r] + bias;
                float g = 0.5f * v * (1.0f + erff(v * 0.70710678118654752f));
                unsigned short gb = f32_to_bf16_rne(g);
                xs_bf[(long)m * H_ + n] = gb;
                int bb = m >> 11, s = m & (S_ - 1);
                xsT_bf[((long)bb * H_ + n) * S_ + s] = gb;
            }
        }
}

// ================= K2: logits = xs @ phi  (WMMA, 2x4 tiles) =================
__global__ __launch_bounds__(32) void logits_kernel(const unsigned short* __restrict__ xs_bf,
                                                    const unsigned short* __restrict__ phiT,
                                                    float* __restrict__ logits) {
    int ntile = blockIdx.x & 1;            // E/64 = 2
    int mtile = blockIdx.x >> 1;           // BS/32 = 1024
    int lane  = threadIdx.x;
    int l15 = lane & 15, kh8 = 8 * (lane >> 4);
    int m0 = mtile * 32, n0 = ntile * 64;
    v8f acc[2][4] = {};
    for (int k0 = 0; k0 < H_; k0 += 32) {
        v16bf a[2], b[4];
#pragma unroll
        for (int i = 0; i < 2; ++i)
            a[i] = load_frag_bf(xs_bf + (long)(m0 + i * 16 + l15) * H_ + k0 + kh8);
#pragma unroll
        for (int j = 0; j < 4; ++j)
            b[j] = load_frag_bf(phiT + (long)(n0 + j * 16 + l15) * H_ + k0 + kh8);
#pragma unroll
        for (int i = 0; i < 2; ++i)
#pragma unroll
            for (int j = 0; j < 4; ++j)
                acc[i][j] = wmma_bf16(a[i], b[j], acc[i][j]);
    }
#pragma unroll
    for (int i = 0; i < 2; ++i)
#pragma unroll
        for (int j = 0; j < 4; ++j) {
            int n = n0 + j * 16 + l15;
#pragma unroll
            for (int r = 0; r < 8; ++r)
                logits[(long)(m0 + i * 16 + kh8 + r) * E_ + n] = acc[i][j][r];
        }
}

// ========== K3: dispatch = softmax over tokens; writes dispT bf16 [B,E,S] ==========
__global__ __launch_bounds__(256) void dispatch_softmax_kernel(const float* __restrict__ logits,
                                                               unsigned short* __restrict__ dispT) {
    int b = blockIdx.x >> 7;               // E_=128
    int e = blockIdx.x & 127;
    const float* col = logits + (long)b * S_ * E_ + e;
    unsigned short* out = dispT + ((long)b * E_ + e) * S_;
    __shared__ float red[256];
    int t = threadIdx.x;

    float mx = -3.402823466e38f;
    for (int m = t; m < S_; m += 256) mx = fmaxf(mx, col[(long)m * E_]);
    red[t] = mx; __syncthreads();
    for (int off = 128; off > 0; off >>= 1) {
        if (t < off) red[t] = fmaxf(red[t], red[t + off]);
        __syncthreads();
    }
    mx = red[0]; __syncthreads();

    float sum = 0.0f;
    for (int m = t; m < S_; m += 256) sum += __expf(col[(long)m * E_] - mx);
    red[t] = sum; __syncthreads();
    for (int off = 128; off > 0; off >>= 1) {
        if (t < off) red[t] += red[t + off];
        __syncthreads();
    }
    float inv = 1.0f / red[0];
    for (int m = t; m < S_; m += 256)
        out[m] = f32_to_bf16_rne(__expf(col[(long)m * E_] - mx) * inv);
}

// ========== K4: combine = softmax over experts; writes combine_bf [BS,E] ==========
__global__ __launch_bounds__(128) void combine_softmax_kernel(const float* __restrict__ logits,
                                                              unsigned short* __restrict__ comb) {
    const float* p = logits + (long)blockIdx.x * E_;
    unsigned short* q = comb + (long)blockIdx.x * E_;
    int t = threadIdx.x;
    __shared__ float red[128];
    float v = p[t];
    red[t] = v; __syncthreads();
    for (int off = 64; off > 0; off >>= 1) {
        if (t < off) red[t] = fmaxf(red[t], red[t + off]);
        __syncthreads();
    }
    float mx = red[0]; __syncthreads();
    float ex = __expf(v - mx);
    red[t] = ex; __syncthreads();
    for (int off = 64; off > 0; off >>= 1) {
        if (t < off) red[t] += red[t + off];
        __syncthreads();
    }
    q[t] = f32_to_bf16_rne(ex / red[0]);
}

// ===== K5: slots[b,e,h] = sum_m dispT[b,e,m]*xsT[b,h,m]  (WMMA NT, 2x4 tiles) =====
// writes slots_bf [B,E,H]
__global__ __launch_bounds__(32) void slots_kernel(const unsigned short* __restrict__ dispT,
                                                   const unsigned short* __restrict__ xsT,
                                                   unsigned short* __restrict__ slots_bf) {
    int htile = blockIdx.x & 3;            // H/64 = 4
    int etile = (blockIdx.x >> 2) & 3;     // E/32 = 4
    int b     = blockIdx.x >> 4;
    const unsigned short* A = dispT + (long)b * E_ * S_;
    const unsigned short* Bm = xsT  + (long)b * H_ * S_;
    int lane = threadIdx.x;
    int l15 = lane & 15, kh8 = 8 * (lane >> 4);
    int e0 = etile * 32, h0 = htile * 64;
    v8f acc[2][4] = {};
    for (int k0 = 0; k0 < S_; k0 += 32) {
        v16bf a[2], bf[4];
#pragma unroll
        for (int i = 0; i < 2; ++i)
            a[i] = load_frag_bf(A + (long)(e0 + i * 16 + l15) * S_ + k0 + kh8);
#pragma unroll
        for (int j = 0; j < 4; ++j)
            bf[j] = load_frag_bf(Bm + (long)(h0 + j * 16 + l15) * S_ + k0 + kh8);
#pragma unroll
        for (int i = 0; i < 2; ++i)
#pragma unroll
            for (int j = 0; j < 4; ++j)
                acc[i][j] = wmma_bf16(a[i], bf[j], acc[i][j]);
    }
#pragma unroll
    for (int i = 0; i < 2; ++i)
#pragma unroll
        for (int j = 0; j < 4; ++j) {
            int h = h0 + j * 16 + l15;
#pragma unroll
            for (int r = 0; r < 8; ++r) {
                int e = e0 + i * 16 + kh8 + r;
                slots_bf[((long)b * E_ + e) * H_ + h] = f32_to_bf16_rne(acc[i][j][r]);
            }
        }
}

// ===== K6: ys[b,e,f] = sum_h slots[b,e,h]*ewT[e,f,h] + eb[e,f] (WMMA, M=B=16, 1x4) =====
// writes ysT_bf [B,H,E]
__global__ __launch_bounds__(32) void expert_kernel(const unsigned short* __restrict__ slots_bf,
                                                    const unsigned short* __restrict__ ewT,
                                                    const float* __restrict__ eb,
                                                    unsigned short* __restrict__ ysT_bf) {
    int ftile = blockIdx.x & 3;            // H/64 = 4
    int e     = blockIdx.x >> 2;           // E_
    const unsigned short* A = slots_bf + (long)e * H_;   // row b, ld = E_*H_
    const unsigned short* Bm = ewT + (long)e * H_ * H_;  // row f, ld = H_
    int lane = threadIdx.x;
    int l15 = lane & 15, kh8 = 8 * (lane >> 4);
    int f0 = ftile * 64;
    v8f acc[4] = {};
    for (int k0 = 0; k0 < H_; k0 += 32) {
        v16bf a = load_frag_bf(A + (long)l15 * (E_ * H_) + k0 + kh8);
        v16bf bf[4];
#pragma unroll
        for (int j = 0; j < 4; ++j)
            bf[j] = load_frag_bf(Bm + (long)(f0 + j * 16 + l15) * H_ + k0 + kh8);
#pragma unroll
        for (int j = 0; j < 4; ++j)
            acc[j] = wmma_bf16(a, bf[j], acc[j]);
    }
#pragma unroll
    for (int j = 0; j < 4; ++j) {
        int f = f0 + j * 16 + l15;
        float bias = eb[(long)e * H_ + f];
#pragma unroll
        for (int r = 0; r < 8; ++r) {
            int b = kh8 + r;               // M index = batch
            ysT_bf[((long)b * H_ + f) * E_ + e] = f32_to_bf16_rne(acc[j][r] + bias);
        }
    }
}

// ===== K7: moe[b,m,h] = sum_e combine[b,m,e]*ysT[b,h,e]  (WMMA NT, 2x4 tiles) =====
// writes moe_bf [BS,H]
__global__ __launch_bounds__(32) void moe_out_kernel(const unsigned short* __restrict__ comb,
                                                     const unsigned short* __restrict__ ysT,
                                                     unsigned short* __restrict__ moe_bf) {
    int htile = blockIdx.x & 3;            // H/64 = 4
    int stile = (blockIdx.x >> 2) & 63;    // S/32 = 64
    int b     = blockIdx.x >> 8;
    const unsigned short* A = comb + (long)b * S_ * E_;
    const unsigned short* Bm = ysT + (long)b * H_ * E_;
    int lane = threadIdx.x;
    int l15 = lane & 15, kh8 = 8 * (lane >> 4);
    int s0 = stile * 32, h0 = htile * 64;
    v8f acc[2][4] = {};
    for (int k0 = 0; k0 < E_; k0 += 32) {
        v16bf a[2], bf[4];
#pragma unroll
        for (int i = 0; i < 2; ++i)
            a[i] = load_frag_bf(A + (long)(s0 + i * 16 + l15) * E_ + k0 + kh8);
#pragma unroll
        for (int j = 0; j < 4; ++j)
            bf[j] = load_frag_bf(Bm + (long)(h0 + j * 16 + l15) * E_ + k0 + kh8);
#pragma unroll
        for (int i = 0; i < 2; ++i)
#pragma unroll
            for (int j = 0; j < 4; ++j)
                acc[i][j] = wmma_bf16(a[i], bf[j], acc[i][j]);
    }
#pragma unroll
    for (int i = 0; i < 2; ++i)
#pragma unroll
        for (int j = 0; j < 4; ++j) {
            int h = h0 + j * 16 + l15;
#pragma unroll
            for (int r = 0; r < 8; ++r) {
                int m = (long)b * S_ + s0 + i * 16 + kh8 + r;
                moe_bf[(long)m * H_ + h] = f32_to_bf16_rne(acc[i][j][r]);
            }
        }
}

// ===== K8: out = x + moe @ fc2_w^T + fc2_b  (WMMA NT, 2x4 tiles) =====
__global__ __launch_bounds__(32) void fc2_res_kernel(const unsigned short* __restrict__ moe_bf,
                                                     const unsigned short* __restrict__ w2b,
                                                     const float* __restrict__ b2,
                                                     const float* __restrict__ x,
                                                     float* __restrict__ out) {
    int ntile = blockIdx.x & 15;           // D/64 = 16
    int mtile = blockIdx.x >> 4;           // BS/32 = 1024
    int lane  = threadIdx.x;
    int l15 = lane & 15, kh8 = 8 * (lane >> 4);
    int m0 = mtile * 32, n0 = ntile * 64;
    v8f acc[2][4] = {};
    for (int k0 = 0; k0 < H_; k0 += 32) {
        v16bf a[2], b[4];
#pragma unroll
        for (int i = 0; i < 2; ++i)
            a[i] = load_frag_bf(moe_bf + (long)(m0 + i * 16 + l15) * H_ + k0 + kh8);
#pragma unroll
        for (int j = 0; j < 4; ++j)
            b[j] = load_frag_bf(w2b + (long)(n0 + j * 16 + l15) * H_ + k0 + kh8);
#pragma unroll
        for (int i = 0; i < 2; ++i)
#pragma unroll
            for (int j = 0; j < 4; ++j)
                acc[i][j] = wmma_bf16(a[i], b[j], acc[i][j]);
    }
#pragma unroll
    for (int i = 0; i < 2; ++i)
#pragma unroll
        for (int j = 0; j < 4; ++j) {
            int n = n0 + j * 16 + l15;
            float bias = b2[n];
#pragma unroll
            for (int r = 0; r < 8; ++r) {
                long idx = (long)(m0 + i * 16 + kh8 + r) * D_ + n;
                out[idx] = x[idx] + acc[i][j][r] + bias;
            }
        }
}

extern "C" void kernel_launch(void* const* d_in, const int* in_sizes, int n_in,
                              void* d_out, int out_size, void* d_ws, size_t ws_size,
                              hipStream_t stream) {
    const float* x   = (const float*)d_in[0];
    const float* w1  = (const float*)d_in[1];
    const float* b1  = (const float*)d_in[2];
    const float* w2  = (const float*)d_in[3];
    const float* b2  = (const float*)d_in[4];
    const float* phi = (const float*)d_in[5];
    const float* ew  = (const float*)d_in[6];
    const float* eb  = (const float*)d_in[7];
    float* out = (float*)d_out;

    // workspace layout (bytes, 256-aligned chunks)
    char* w = (char*)d_ws;
    auto alloc = [&](size_t bytes) {
        char* p = w;
        w += (bytes + 255) & ~(size_t)255;
        return p;
    };
    unsigned short* x_bf     = (unsigned short*)alloc((size_t)BS_ * D_ * 2);    // 64 MB
    unsigned short* w1_bf    = (unsigned short*)alloc((size_t)H_ * D_ * 2);
    unsigned short* w2_bf    = (unsigned short*)alloc((size_t)D_ * H_ * 2);
    unsigned short* phiT     = (unsigned short*)alloc((size_t)E_ * H_ * 2);
    unsigned short* ewT      = (unsigned short*)alloc((size_t)E_ * H_ * H_ * 2);// 16 MB
    unsigned short* xs_bf    = (unsigned short*)alloc((size_t)BS_ * H_ * 2);    // 16 MB
    unsigned short* xsT_bf   = (unsigned short*)alloc((size_t)B_ * H_ * S_ * 2);// 16 MB
    float*          logits   = (float*)        alloc((size_t)BS_ * E_ * 4);    // 16 MB
    unsigned short* comb_bf  = (unsigned short*)alloc((size_t)BS_ * E_ * 2);    //  8 MB
    unsigned short* dispT_bf = (unsigned short*)alloc((size_t)B_ * E_ * S_ * 2);//  8 MB
    unsigned short* slots_bf = (unsigned short*)alloc((size_t)B_ * E_ * H_ * 2);
    unsigned short* ysT_bf   = (unsigned short*)alloc((size_t)B_ * H_ * E_ * 2);
    unsigned short* moe_bf   = (unsigned short*)alloc((size_t)BS_ * H_ * 2);    // 16 MB

    // --- operand preparation (bandwidth passes) ---
    cvt_bf16_kernel<<<((size_t)BS_ * D_ / 2 + 255) / 256, 256, 0, stream>>>(x,  x_bf,  (long)BS_ * D_ / 2);
    cvt_bf16_kernel<<<((size_t)H_  * D_ / 2 + 255) / 256, 256, 0, stream>>>(w1, w1_bf, (long)H_  * D_ / 2);
    cvt_bf16_kernel<<<((size_t)D_  * H_ / 2 + 255) / 256, 256, 0, stream>>>(w2, w2_bf, (long)D_  * H_ / 2);
    phiT_kernel<<<(H_ * E_) / 256, 256, 0, stream>>>(phi, phiT);
    ewT_kernel<<<(long)E_ * H_ * H_ / 256, 256, 0, stream>>>(ew, ewT);

    // --- pipeline ---
    fc1_gelu_kernel       <<<(BS_/32)*(H_/64), 32, 0, stream>>>(x_bf, w1_bf, b1, xs_bf, xsT_bf);
    logits_kernel         <<<(BS_/32)*(E_/64), 32, 0, stream>>>(xs_bf, phiT, logits);
    dispatch_softmax_kernel<<<B_*E_,          256, 0, stream>>>(logits, dispT_bf);
    combine_softmax_kernel<<<BS_,             128, 0, stream>>>(logits, comb_bf);
    slots_kernel          <<<B_*(E_/32)*(H_/64), 32, 0, stream>>>(dispT_bf, xsT_bf, slots_bf);
    expert_kernel         <<<E_*(H_/64),      32, 0, stream>>>(slots_bf, ewT, eb, ysT_bf);
    moe_out_kernel        <<<B_*(S_/32)*(H_/64), 32, 0, stream>>>(comb_bf, ysT_bf, moe_bf);
    fc2_res_kernel        <<<(BS_/32)*(D_/64), 32, 0, stream>>>(moe_bf, w2_bf, b2, x, out);
}
